// EntEncoderFast_46102178955797
// MI455X (gfx1250) — matrix operand order repaired and checked by reading, not compiled, not run on hardware
//
#include <hip/hip_runtime.h>
#include <hip/hip_bf16.h>

typedef __attribute__((ext_vector_type(16))) _Float16 v16h;
typedef __attribute__((ext_vector_type(8)))  _Float16 v8h;
typedef __attribute__((ext_vector_type(8)))  float    v8f;

#define GG 64
#define HH 32
#define WW 64

// ---------------------------------------------------------------------------
// Fold the autoregressive group mask into the weights, convert f32 -> f16,
// repack to [tap][kchunk][cout][kk32] so B-tile staging is contiguous.
// w layout: [Cout][Cin][5][5]  (idx = (co*Cin+ci)*25 + tap)
// ---------------------------------------------------------------------------
__global__ __launch_bounds__(256) void repack_weights(
    const float* __restrict__ w, _Float16* __restrict__ Wp,
    int Cout, int Cin, int div_o, int div_i, int strict)
{
    int idx = blockIdx.x * 256 + threadIdx.x;
    int total = Cout * Cin * 25;
    if (idx >= total) return;
    int tap  = idx % 25;
    int rest = idx / 25;
    int ci   = rest % Cin;
    int co   = rest / Cin;
    int ky = tap / 5, kx = tap % 5;
    int tot = (ci / div_i - co / div_o) + (ky - 2) + (kx - 2);
    bool keep = strict ? (tot < 0) : (tot <= 0);
    float v = keep ? w[idx] : 0.0f;
    int kcN = Cin >> 5;
    size_t o = ((size_t)(tap * kcN + (ci >> 5)) * Cout + co) * 32 + (ci & 31);
    Wp[o] = (_Float16)v;
}

// ---------------------------------------------------------------------------
// tdata = (data - 3.5) * mask, f32 -> f16, NCHW -> [h][w][c] channels-last.
// (Batch of 3 in the reference is redundant: all entries identical.)
// ---------------------------------------------------------------------------
__global__ __launch_bounds__(256) void prep_input(
    const float* __restrict__ data, const float* __restrict__ maskp,
    _Float16* __restrict__ X0)
{
    int i = blockIdx.x * 256 + threadIdx.x;          // over G*H*W, g-major
    if (i >= GG * HH * WW) return;
    int g = i / (HH * WW);
    int r = i % (HH * WW);
    int h = r / WW, w = r % WW;
    float v = (data[i] - 3.5f) * maskp[i];
    X0[((size_t)(h * WW + w)) * GG + g] = (_Float16)v;
}

// ---------------------------------------------------------------------------
// Implicit-GEMM masked conv via v_wmma_f32_16x16x32_f16.
// Workgroup = (row h, 64-cout tile); 8 wave32s; output 64 spatial x 64 cout.
// 64 K-channels staged per barrier round -> 4 WMMAs per wave per round.
// LDS rows padded to 72 halves (144B = 9*16B) for bank spread + alignment.
// Epilogue transposes through LDS for coalesced b128 stores (f16 path).
// ---------------------------------------------------------------------------
template<int KCN, bool RELU, bool HASRES, bool F32OUT>
__global__ __launch_bounds__(256) void conv_wmma(
    const _Float16* __restrict__ Xin,   // [h][w][CIN] f16
    const _Float16* __restrict__ Wp,    // repacked weights
    const float*    __restrict__ bias,  // [Cout]
    const _Float16* __restrict__ Res,   // optional residual [h][w][Cout]
    _Float16*       __restrict__ Xout,  // f16 out [h][w][Cout]
    float*          __restrict__ Yout,  // f32 out (final layer)
    int Cout)
{
    constexpr int CIN = KCN * 32;
    __shared__ __align__(16) _Float16 As[64][72];
    __shared__ __align__(16) _Float16 Bs[64][72];

    const int h    = blockIdx.x;        // output row
    const int co0  = blockIdx.y * 64;   // cout tile base
    const int t    = threadIdx.x;
    const int lane = t & 31;
    const int wv   = t >> 5;            // wave id 0..7
    const int m16  = wv & 3;            // M (spatial) 16-block
    const int n16  = (wv >> 2) * 2;     // first of two N 16-blocks
    const int rr   = lane & 15;
    const int hb   = lane >> 4;         // lane half (K split per ISA layout)
    const int wl   = t >> 2;            // staging row 0..63
    const int e8   = (t & 3) * 8;       // staging 8-half chunk

    // Loop-invariant LDS fragment row pointers.
    const _Float16* aRow  = &As[m16 * 16 + rr][0];
    const _Float16* bRow0 = &Bs[(n16 + 0) * 16 + rr][0];
    const _Float16* bRow1 = &Bs[(n16 + 1) * 16 + rr][0];

    v8f acc0 = {};
    v8f acc1 = {};

    for (int ky = 0; ky < 5; ++ky) {
        const int hs = h + ky - 2;
        const bool hok = (hs >= 0) && (hs < HH);
        for (int kx = 0; kx < 5; ++kx) {
            const int tap = ky * 5 + kx;
            const int ws  = wl + kx - 2;
            const bool aok = hok && (ws >= 0) && (ws < WW);
            const _Float16* xrow = Xin + (size_t)(hs * WW + ws) * CIN;
            const _Float16* wtap = Wp + (size_t)tap * KCN * Cout * 32;
            for (int kc0 = 0; kc0 < KCN; kc0 += 2) {
                // ---- stage A and B: two 32-channel chunks per round ----
                #pragma unroll
                for (int c2 = 0; c2 < 2; ++c2) {
                    v8h av = {};
                    if (aok)
                        av = *(const v8h*)(xrow + (kc0 + c2) * 32 + e8);
                    *(v8h*)(&As[wl][c2 * 32 + e8]) = av;
                    *(v8h*)(&Bs[wl][c2 * 32 + e8]) =
                        *(const v8h*)(wtap + ((size_t)(kc0 + c2) * Cout + co0 + wl) * 32 + e8);
                }
                if (kc0 + 2 < KCN)   // prefetch next weight chunk (global_prefetch_b8)
                    __builtin_prefetch(
                        wtap + ((size_t)(kc0 + 2) * Cout + co0 + wl) * 32 + e8, 0, 3);
                __syncthreads();

                // ---- 2 K-steps of WMMA per round (shared A per N pair) ----
                #pragma unroll
                for (int c2 = 0; c2 < 2; ++c2) {
                    v16h a, b0, b1;
                    {
                        v8h lo = *(const v8h*)(aRow + c2 * 32 + hb * 8);
                        v8h hi = *(const v8h*)(aRow + c2 * 32 + 16 + hb * 8);
                        #pragma unroll
                        for (int i = 0; i < 8; ++i) { a[i] = lo[i]; a[8 + i] = hi[i]; }
                    }
                    {
                        v8h lo = *(const v8h*)(bRow0 + c2 * 32 + hb * 8);
                        v8h hi = *(const v8h*)(bRow0 + c2 * 32 + 16 + hb * 8);
                        #pragma unroll
                        for (int i = 0; i < 8; ++i) { b0[i] = lo[i]; b0[8 + i] = hi[i]; }
                    }
                    {
                        v8h lo = *(const v8h*)(bRow1 + c2 * 32 + hb * 8);
                        v8h hi = *(const v8h*)(bRow1 + c2 * 32 + 16 + hb * 8);
                        #pragma unroll
                        for (int i = 0; i < 8; ++i) { b1[i] = lo[i]; b1[8 + i] = hi[i]; }
                    }
                    acc0 = __builtin_amdgcn_wmma_f32_16x16x32_f16(
                        false, a, false, b0, (short)0, acc0, false, false);
                    acc1 = __builtin_amdgcn_wmma_f32_16x16x32_f16(
                        false, a, false, b1, (short)0, acc1, false, false);
                }
                __syncthreads();
            }
        }
    }

    // ---- epilogue ----
    const int n0l = (n16 + 0) * 16 + rr;
    const int n1l = (n16 + 1) * 16 + rr;
    const float bv0 = bias[co0 + n0l];
    const float bv1 = bias[co0 + n1l];

    if (F32OUT) {
        // Final layer: direct f32 stores (1 of 12 layers, cost negligible).
        #pragma unroll
        for (int j = 0; j < 8; ++j) {
            const int wpos = m16 * 16 + hb * 8 + j;   // C/D layout: M=j (+8 hi lanes)
            const size_t base = (size_t)(h * WW + wpos) * Cout;
            float v0 = acc0[j] + bv0;
            float v1 = acc1[j] + bv1;
            if (RELU) { v0 = fmaxf(v0, 0.0f); v1 = fmaxf(v1, 0.0f); }
            Yout[base + co0 + n0l] = v0;
            Yout[base + co0 + n1l] = v1;
        }
    } else {
        // Transpose through LDS so global stores are coalesced b128s.
        #pragma unroll
        for (int j = 0; j < 8; ++j) {
            const int wpos = m16 * 16 + hb * 8 + j;
            float v0 = acc0[j] + bv0;
            float v1 = acc1[j] + bv1;
            if (RELU) { v0 = fmaxf(v0, 0.0f); v1 = fmaxf(v1, 0.0f); }
            As[wpos][n0l] = (_Float16)v0;
            As[wpos][n1l] = (_Float16)v1;
        }
        __syncthreads();
        const int c16 = (t & 3) * 16;
        _Float16* orow = Xout + (size_t)(h * WW + wl) * Cout + co0 + c16;
        v8h o0 = *(const v8h*)(&As[wl][c16]);
        v8h o1 = *(const v8h*)(&As[wl][c16 + 8]);
        if (HASRES) {
            const _Float16* rrow = Res + (size_t)(h * WW + wl) * Cout + co0 + c16;
            v8h r0 = *(const v8h*)(rrow);
            v8h r1 = *(const v8h*)(rrow + 8);
            o0 = o0 + r0;
            o1 = o1 + r1;
        }
        *(v8h*)(orow)     = o0;
        *(v8h*)(orow + 8) = o1;
    }
}

// ---------------------------------------------------------------------------
// Degenerate mixture (identical components -> softmax weights sum to 1):
// table[g,h,w,k] = 0.5*(1+erf((edge_k - mu)/(sig*sqrt2))) * 65536
// Y layout: [h][w][192] f32, channel 3g+{0:logit,1:mu,2:sig-raw}.
// ---------------------------------------------------------------------------
__global__ __launch_bounds__(256) void mixture_table(
    const float* __restrict__ Y, float* __restrict__ out)
{
    int i = blockIdx.x * 256 + threadIdx.x;          // over G*H*W, g-major
    if (i >= GG * HH * WW) return;
    int g = i / (HH * WW);
    int r = i % (HH * WW);
    int h = r / WW, w = r % WW;
    const float* yb = Y + (size_t)(h * WW + w) * 192 + g * 3;
    float mu = yb[1];
    float sv = yb[2];
    float sig = (sv > 20.0f ? sv : log1pf(expf(sv))) + 1e-6f;
    float inv = 0.70710678118f / sig;                // 1/(sig*sqrt(2))
    #pragma unroll
    for (int k = 0; k < 8; ++k) {
        float edge = (float)k + 0.5f - 3.5f;
        float z = (edge - mu) * inv;
        float cdf = 0.5f * (1.0f + erff(z));
        out[(size_t)i * 8 + k] = cdf * 65536.0f;
    }
}

// ---------------------------------------------------------------------------
extern "C" void kernel_launch(void* const* d_in, const int* in_sizes, int n_in,
                              void* d_out, int out_size, void* d_ws, size_t ws_size,
                              hipStream_t stream)
{
    (void)in_sizes; (void)n_in; (void)out_size; (void)ws_size;
    const float* data  = (const float*)d_in[0];
    const float* maskp = (const float*)d_in[1];
    const float* w_in  = (const float*)d_in[2];
    const float* b_in  = (const float*)d_in[3];
    const float* w_hid = (const float*)d_in[4];
    const float* b_hid = (const float*)d_in[5];
    const float* w_out = (const float*)d_in[6];
    const float* b_out = (const float*)d_in[7];

    char* ws = (char*)d_ws;
    size_t off = 0;
    auto alloc = [&](size_t bytes) {
        void* p = ws + off;
        off += (bytes + 255) & ~(size_t)255;
        return p;
    };

    const size_t wpInElems  = (size_t)25 * 2 * 256 * 32;   // Cin=64
    const size_t wpHidElems = (size_t)25 * 8 * 256 * 32;   // Cin=256
    const size_t wpOutElems = (size_t)25 * 8 * 192 * 32;

    _Float16* WpIn  = (_Float16*)alloc(wpInElems  * sizeof(_Float16));
    _Float16* WpHid = (_Float16*)alloc(10 * wpHidElems * sizeof(_Float16));
    _Float16* WpOut = (_Float16*)alloc(wpOutElems * sizeof(_Float16));
    _Float16* X0 = (_Float16*)alloc((size_t)HH * WW * 64  * sizeof(_Float16));
    _Float16* Xa = (_Float16*)alloc((size_t)HH * WW * 256 * sizeof(_Float16));
    _Float16* Xb = (_Float16*)alloc((size_t)HH * WW * 256 * sizeof(_Float16));
    _Float16* Xc = (_Float16*)alloc((size_t)HH * WW * 256 * sizeof(_Float16));
    float*    Yf = (float*)   alloc((size_t)HH * WW * 192 * sizeof(float));

    // ---- weight repack (mask folded, f16) ----
    repack_weights<<<(256 * 64 * 25 + 255) / 256, 256, 0, stream>>>(
        w_in, WpIn, 256, 64, 4, 1, 1);
    for (int l = 0; l < 10; ++l) {
        repack_weights<<<(256 * 256 * 25 + 255) / 256, 256, 0, stream>>>(
            w_hid + (size_t)l * 256 * 256 * 25, WpHid + (size_t)l * wpHidElems,
            256, 256, 4, 4, 0);
    }
    repack_weights<<<(192 * 256 * 25 + 255) / 256, 256, 0, stream>>>(
        w_out, WpOut, 192, 256, 3, 4, 0);

    // ---- input transform (single batch; reference batch-of-3 is redundant) ----
    prep_input<<<(GG * HH * WW + 255) / 256, 256, 0, stream>>>(data, maskp, X0);

    // ---- conv chain ----
    dim3 g256(HH, 4), blk(256);
    conv_wmma<2, true, false, false><<<g256, blk, 0, stream>>>(
        X0, WpIn, b_in, nullptr, Xa, nullptr, 256);
    for (int i = 0; i < 5; ++i) {
        conv_wmma<8, true, false, false><<<g256, blk, 0, stream>>>(
            Xa, WpHid + (size_t)(2 * i) * wpHidElems, b_hid + (size_t)(2 * i) * 256,
            nullptr, Xb, nullptr, 256);
        conv_wmma<8, true, true, false><<<g256, blk, 0, stream>>>(
            Xb, WpHid + (size_t)(2 * i + 1) * wpHidElems, b_hid + (size_t)(2 * i + 1) * 256,
            Xa, Xc, nullptr, 256);
        _Float16* tmp = Xa; Xa = Xc; Xc = tmp;
    }
    dim3 g192(HH, 3);
    conv_wmma<8, false, false, true><<<g192, blk, 0, stream>>>(
        Xa, WpOut, b_out, nullptr, nullptr, Yf, 192);

    // ---- CDF table ----
    mixture_table<<<(GG * HH * WW + 255) / 256, 256, 0, stream>>>(Yf, (float*)d_out);
}
